// MultiHeadedAttention_72121090834744
// MI455X (gfx1250) — matrix operand order repaired and checked by reading, not compiled, not run on hardware
//
#include <hip/hip_runtime.h>

// MHA forward for B=4, S=2048, D=1024, H=16, DH=64 on gfx1250 (MI455X).
// (1) weights -> bf16 pre-swizzled B-fragment order, (2) activations -> bf16,
// (3) QKV bf16 WMMA GEMMs (Q pre-scaled by 0.125*log2e; V transposed [B,H,DH,S]),
// (4) flash attention, 64-key tiles: K/V staged to LDS via async-DMA
//     (GLOBAL_LOAD_ASYNC_TO_LDS_B128, double-buffered, once per block instead of
//     once per wave), WMMA QK^T / PV, fused-DPP exp2-domain online softmax,
// (5) bf16 WMMA output projection -> fp32.
// Workspace: 8MB weights + 48MB bf16 acts + 48MB Q/K/V = 104MB (Ctx aliases acts).

typedef __attribute__((ext_vector_type(16))) __bf16 v16bf;
typedef __attribute__((ext_vector_type(8)))  float v8f;
typedef __attribute__((ext_vector_type(4)))  int   v4i;

typedef __attribute__((address_space(1))) v4i* g4_ptr;   // global int4*
typedef __attribute__((address_space(3))) v4i* l4_ptr;   // LDS int4*

union Frag16 {
  uint4 u4[2];
  unsigned short s[16];
  v16bf v;
};

__device__ __forceinline__ unsigned short f2bf(float f) {
  union { float f; unsigned u; } x; x.f = f;
  unsigned r = x.u + 0x7FFFu + ((x.u >> 16) & 1u);   // round-to-nearest-even
  return (unsigned short)(r >> 16);
}

__device__ __forceinline__ float fast_exp2(float x) { return __builtin_amdgcn_exp2f(x); }

// Async global->LDS 16-byte copy (ASYNCcnt-tracked DMA path).
__device__ __forceinline__ void async_b128(const unsigned short* g, unsigned short* l) {
#if __has_builtin(__builtin_amdgcn_global_load_async_to_lds_b128)
  __builtin_amdgcn_global_load_async_to_lds_b128((g4_ptr)g, (l4_ptr)l, 0, 0);
#else
  unsigned loff = (unsigned)(__SIZE_TYPE__)(__attribute__((address_space(3))) void*)l;
  asm volatile("global_load_async_to_lds_b128 %0, %1, off"
               :: "v"(loff), "v"(g) : "memory");
#endif
}
__device__ __forceinline__ void wait_async0() {
#if __has_builtin(__builtin_amdgcn_s_wait_asynccnt)
  __builtin_amdgcn_s_wait_asynccnt(0);
#else
  asm volatile("s_wait_asynccnt 0" ::: "memory");
#endif
}

// Fused DPP butterfly reduction over each 16-lane row group: one VALU per step.
#define DPP_MAX(x, CTRL)                                                          \
  { float _o;                                                                     \
    asm("v_max_num_f32_dpp %0, %1, %1 " CTRL                                      \
        " row_mask:0xf bank_mask:0xf bound_ctrl:1" : "=v"(_o) : "v"(x));          \
    x = _o; }
#define DPP_ADD(x, CTRL)                                                          \
  { float _o;                                                                     \
    asm("v_add_f32_dpp %0, %1, %1 " CTRL                                          \
        " row_mask:0xf bank_mask:0xf bound_ctrl:1" : "=v"(_o) : "v"(x));          \
    x = _o; }

__device__ __forceinline__ float red_max16(float v) {
  DPP_MAX(v, "quad_perm:[1,0,3,2]")
  DPP_MAX(v, "quad_perm:[2,3,0,1]")
  DPP_MAX(v, "row_half_mirror")
  DPP_MAX(v, "row_mirror")
  return v;
}
__device__ __forceinline__ float red_sum16(float v) {
  DPP_ADD(v, "quad_perm:[1,0,3,2]")
  DPP_ADD(v, "quad_perm:[2,3,0,1]")
  DPP_ADD(v, "row_half_mirror")
  DPP_ADD(v, "row_mirror")
  return v;
}
// Plain max in asm (avoids compiler re-canonicalizing asm-produced values).
__device__ __forceinline__ float asm_max(float a, float b) {
  float o;
  asm("v_max_num_f32 %0, %1, %2" : "=v"(o) : "v"(a), "v"(b));
  return o;
}

constexpr int B_ = 4, S_ = 2048, D_ = 1024, H_ = 16, DH_ = 64;

// ---------------------------------------------------------------------------
// fp32 weight [D][D] -> bf16 in per-lane B-fragment order:
// dst[(((kt*64+nt)*32+lane)*16+i)] = W[kt*32 + i + 16*(lane/16)][nt*16 + lane%16]
// ---------------------------------------------------------------------------
__global__ void cvt_weights_k(const float* __restrict__ Wq, const float* __restrict__ Wk,
                              const float* __restrict__ Wv, const float* __restrict__ Wo,
                              unsigned short* __restrict__ dst) {
  const int e = blockIdx.x * 256 + threadIdx.x;       // 0 .. D*D-1 (dst order)
  const int which = blockIdx.y;
  const float* src = (which == 0) ? Wq : (which == 1) ? Wk : (which == 2) ? Wv : Wo;
  unsigned short* d = dst + (size_t)which * D_ * D_;
  const int i    = e & 15;
  const int lane = (e >> 4) & 31;
  const int nt   = (e >> 9) & 63;
  const int kt   = e >> 15;
  const int k = kt * 32 + i + ((lane >> 4) << 4);
  const int n = nt * 16 + (lane & 15);
  d[e] = f2bf(src[(size_t)k * D_ + n]);
}

// ---------------------------------------------------------------------------
// fp32 activations (query/key/value) -> bf16 row-major, 8 elements/thread.
// ---------------------------------------------------------------------------
__global__ void cvt_act_k(const float* __restrict__ q, const float* __restrict__ k,
                          const float* __restrict__ v, unsigned short* __restrict__ dst) {
  const int which = blockIdx.y;
  const float* src = (which == 0) ? q : (which == 1) ? k : v;
  const size_t base = ((size_t)blockIdx.x * 256 + threadIdx.x) * 8;
  unsigned short* d = dst + (size_t)which * ((size_t)B_ * S_ * D_) + base;
  float4 f0 = *(const float4*)(src + base);
  float4 f1 = *(const float4*)(src + base + 4);
  uint4 pk;
  pk.x = (unsigned)f2bf(f0.x) | ((unsigned)f2bf(f0.y) << 16);
  pk.y = (unsigned)f2bf(f0.z) | ((unsigned)f2bf(f0.w) << 16);
  pk.z = (unsigned)f2bf(f1.x) | ((unsigned)f2bf(f1.y) << 16);
  pk.w = (unsigned)f2bf(f1.z) | ((unsigned)f2bf(f1.w) << 16);
  *(uint4*)d = pk;
}

// ---------------------------------------------------------------------------
// GEMM: out[M=8192, N=1024] = A_bf16[M,1024] x W_swizzled + bias, then *ascale.
// MODE 0: bf16 out [B,H,S,DH] (Q,K) | MODE 1: bf16 out [B,H,DH,S] (V) |
// MODE 2: fp32 out [M,N] (final projection, ascale unused).
// ---------------------------------------------------------------------------
template <int MODE>
__global__ void gemm_wmma_k(const unsigned short* __restrict__ A,
                            const unsigned short* __restrict__ Wsw,
                            const float* __restrict__ bias,
                            void* __restrict__ outp, float ascale) {
  const int lane = threadIdx.x & 31;
  const int wave = threadIdx.x >> 5;
  const int wm = wave & 3, wn = wave >> 2;
  const int m0 = blockIdx.x * 64 + wm * 16;
  const int n0 = blockIdx.y * 128 + wn * 64;
  const int hl = lane >> 4, l16 = lane & 15;

  v8f acc[4] = {};
  const unsigned short* arow = A + (size_t)(m0 + l16) * D_ + 16 * hl;

  for (int kb = 0; kb < D_; kb += 32) {
    Frag16 af;
    af.u4[0] = *(const uint4*)(arow + kb);
    af.u4[1] = *(const uint4*)(arow + kb + 8);
    if (kb + 32 < D_) __builtin_prefetch(arow + kb + 32, 0, 1);   // global_prefetch_b8
    const int kt = kb >> 5;
#pragma unroll
    for (int t = 0; t < 4; ++t) {
      const int nt = (n0 >> 4) + t;
      Frag16 bf_;
      const uint4* bp = (const uint4*)(Wsw + ((((size_t)kt * 64 + nt) * 32 + lane) << 4));
      bf_.u4[0] = bp[0];
      bf_.u4[1] = bp[1];
      acc[t] = __builtin_amdgcn_wmma_f32_16x16x32_bf16(false, af.v, false, bf_.v,
                                                       (short)0, acc[t], false, false);
    }
  }

  float*          outF = (float*)outp;
  unsigned short* outB = (unsigned short*)outp;
#pragma unroll
  for (int t = 0; t < 4; ++t) {
    const int n = n0 + t * 16 + l16;
    const float bv = bias[n];
    if (MODE == 2) {
#pragma unroll
      for (int r = 0; r < 8; ++r) {
        const int m = m0 + r + 8 * hl;
        outF[(size_t)m * D_ + n] = acc[t][r] + bv;
      }
    } else if (MODE == 0) {            // [B,H,S,DH] bf16
      const int hh = n >> 6, dh = n & 63;
#pragma unroll
      for (int r = 0; r < 8; ++r) {
        const int m = m0 + r + 8 * hl;
        const int b = m >> 11, s = m & 2047;
        outB[((((size_t)b * H_ + hh) * S_ + s) << 6) + dh] = f2bf((acc[t][r] + bv) * ascale);
      }
    } else {                           // MODE 1: [B,H,DH,S] bf16, packed 8-wide store
      const int hh = n >> 6, dh = n & 63;
      const int b = m0 >> 11;
      const int sBase = (m0 & 2047) + 8 * hl;
      unsigned short v8[8];
#pragma unroll
      for (int r = 0; r < 8; ++r) v8[r] = f2bf((acc[t][r] + bv) * ascale);
      uint4 pk;
      pk.x = (unsigned)v8[0] | ((unsigned)v8[1] << 16);
      pk.y = (unsigned)v8[2] | ((unsigned)v8[3] << 16);
      pk.z = (unsigned)v8[4] | ((unsigned)v8[5] << 16);
      pk.w = (unsigned)v8[6] | ((unsigned)v8[7] << 16);
      *(uint4*)&outB[(((size_t)b * H_ + hh) * DH_ + dh) * (size_t)S_ + sBase] = pk;
    }
  }
}

// ---------------------------------------------------------------------------
// Flash attention. Block = 8 waves, one (b,h), 128 query rows (16 per wave).
// Per 64-key tile: K (8KB) + V (8KB) async-DMA'd into LDS once per block,
// double-buffered; QK^T (8 WMMA) from LDS fragments, one exp2-domain online
// softmax with fused-DPP reductions, P via per-wave LDS bounce, PV (8 WMMA).
// ---------------------------------------------------------------------------
__global__ void attention_k(const unsigned short* __restrict__ Qb,
                            const unsigned short* __restrict__ Kb,
                            const unsigned short* __restrict__ Vt,
                            unsigned short* __restrict__ Ctx) {
  __shared__ __align__(16) unsigned short kbuf[2][64 * 64];   // 2 x 8 KB, [key][dh]
  __shared__ __align__(16) unsigned short vbuf[2][64 * 64];   // 2 x 8 KB, [dh][key]
  __shared__ __align__(16) unsigned short pbuf[8][16 * 64];   // 16 KB, per-wave P
  const int tid  = threadIdx.x;
  const int lane = tid & 31;
  const int wave = tid >> 5;
  const int hl = lane >> 4, l16 = lane & 15;
  const int q0 = blockIdx.x * 128 + wave * 16;
  const int bh = blockIdx.y;
  const int bidx = bh >> 4, h = bh & 15;

  const unsigned short* Kbase = Kb + (size_t)bh * S_ * DH_;   // [S][64]
  const unsigned short* Vbase = Vt + (size_t)bh * DH_ * S_;   // [64][S]

  // Stage one 64-key tile: K tile is a contiguous 8KB slab; V tile is 64
  // strided 128B rows. 4 async b128 per thread per tile.
  auto stage = [&](int buf, int key0) {
    const unsigned short* kg = Kbase + (size_t)key0 * DH_;
    async_b128(kg + tid * 8,         &kbuf[buf][tid * 8]);
    async_b128(kg + (tid + 256) * 8, &kbuf[buf][(tid + 256) * 8]);
#pragma unroll
    for (int j = 0; j < 2; ++j) {
      const int idx = tid + j * 256;             // 0..511
      const int dh = idx >> 3, ch = idx & 7;
      async_b128(Vbase + (size_t)dh * S_ + key0 + ch * 8, &vbuf[buf][dh * 64 + ch * 8]);
    }
  };

  Frag16 qf[2];
  {
    const unsigned short* qp = Qb + ((size_t)bh * S_ + (q0 + l16)) * DH_;
#pragma unroll
    for (int ks = 0; ks < 2; ++ks) {
      const unsigned short* p = qp + ks * 32 + 8 * hl;
      qf[ks].u4[0] = *(const uint4*)(p);
      qf[ks].u4[1] = *(const uint4*)(p + 16);
    }
  }

  float mrow[8], lrow[8];
#pragma unroll
  for (int r = 0; r < 8; ++r) { mrow[r] = -1e30f; lrow[r] = 0.0f; }
  v8f o[4] = {};

  unsigned short* smw = &pbuf[wave][0];

  stage(0, 0);
  for (int it = 0; it < S_ / 64; ++it) {
    const int cur = it & 1;
    wait_async0();
    __syncthreads();                       // tile[cur] resident for all waves
    if (it + 1 < S_ / 64) stage(cur ^ 1, (it + 1) * 64);

    v8f sc[4] = {};
#pragma unroll
    for (int sub = 0; sub < 4; ++sub) {
      const unsigned short* kp = &kbuf[cur][(sub * 16 + l16) * 64 + 8 * hl];
#pragma unroll
      for (int ks = 0; ks < 2; ++ks) {
        Frag16 kf;
        const unsigned short* p = kp + ks * 32;
        kf.u4[0] = *(const uint4*)(p);
        kf.u4[1] = *(const uint4*)(p + 16);
        sc[sub] = __builtin_amdgcn_wmma_f32_16x16x32_bf16(false, qf[ks].v, false, kf.v,
                                                          (short)0, sc[sub], false, false);
      }
    }
    // one online-softmax pass per 64 keys (scores already exp2-domain logits)
#pragma unroll
    for (int r = 0; r < 8; ++r) {
      const float s0 = sc[0][r], s1 = sc[1][r], s2 = sc[2][r], s3 = sc[3][r];
      float mx = fmaxf(fmaxf(s0, s1), fmaxf(s2, s3));
      mx = red_max16(mx);
      const float mnew  = asm_max(mrow[r], mx);
      const float alpha = fast_exp2(mrow[r] - mnew);
      mrow[r] = mnew;
      const float p0 = fast_exp2(s0 - mnew);
      const float p1 = fast_exp2(s1 - mnew);
      const float p2 = fast_exp2(s2 - mnew);
      const float p3 = fast_exp2(s3 - mnew);
      float rs = (p0 + p1) + (p2 + p3);
      rs = red_sum16(rs);
      lrow[r] = lrow[r] * alpha + rs;
#pragma unroll
      for (int t = 0; t < 4; ++t) o[t][r] *= alpha;
      const int rowoff = (r + 8 * hl) * 64 + l16;
      smw[rowoff]      = f2bf(p0);
      smw[rowoff + 16] = f2bf(p1);
      smw[rowoff + 32] = f2bf(p2);
      smw[rowoff + 48] = f2bf(p3);
    }
    asm volatile("" ::: "memory");  // same-wave LDS is in-order; compiler fence only
#pragma unroll
    for (int half = 0; half < 2; ++half) {
      Frag16 pf;
      const unsigned short* p = smw + l16 * 64 + half * 32 + 8 * hl;
      pf.u4[0] = *(const uint4*)(p);
      pf.u4[1] = *(const uint4*)(p + 16);
#pragma unroll
      for (int t = 0; t < 4; ++t) {
        Frag16 vf;
        const unsigned short* vp = &vbuf[cur][(t * 16 + l16) * 64 + half * 32 + 8 * hl];
        vf.u4[0] = *(const uint4*)(vp);
        vf.u4[1] = *(const uint4*)(vp + 16);
        o[t] = __builtin_amdgcn_wmma_f32_16x16x32_bf16(false, pf.v, false, vf.v,
                                                       (short)0, o[t], false, false);
      }
    }
    asm volatile("" ::: "memory");
    // No trailing barrier needed: next overwrite of tile[cur] is issued only
    // after every wave passes the next iteration's barrier (double buffering).
  }

  float inv[8];
#pragma unroll
  for (int r = 0; r < 8; ++r) inv[r] = 1.0f / lrow[r];
#pragma unroll
  for (int t = 0; t < 4; ++t) {
#pragma unroll
    for (int r = 0; r < 8; ++r) {
      const int row = q0 + r + 8 * hl;
      Ctx[((size_t)bidx * S_ + row) * D_ + h * DH_ + t * 16 + l16] = f2bf(o[t][r] * inv[r]);
    }
  }
}

// ---------------------------------------------------------------------------
extern "C" void kernel_launch(void* const* d_in, const int* in_sizes, int n_in,
                              void* d_out, int out_size, void* d_ws, size_t ws_size,
                              hipStream_t stream) {
  (void)in_sizes; (void)n_in; (void)out_size; (void)ws_size;
  const float* query = (const float*)d_in[0];
  const float* key   = (const float*)d_in[1];
  const float* value = (const float*)d_in[2];
  const float* Wq = (const float*)d_in[3];
  const float* bq = (const float*)d_in[4];
  const float* Wk = (const float*)d_in[5];
  const float* bk = (const float*)d_in[6];
  const float* Wv = (const float*)d_in[7];
  const float* bv = (const float*)d_in[8];
  const float* Wo = (const float*)d_in[9];
  const float* bo = (const float*)d_in[10];

  char* ws = (char*)d_ws;
  const size_t WW  = (size_t)D_ * D_ * sizeof(unsigned short);        // 2 MB
  const size_t ACT = (size_t)B_ * S_ * D_ * sizeof(unsigned short);   // 16 MB
  unsigned short* Wsw  = (unsigned short*)ws;                         // 4 * 2 MB
  unsigned short* Acts = (unsigned short*)(ws + 4 * WW);              // 3 * 16 MB
  unsigned short* Qb   = (unsigned short*)(ws + 4 * WW + 3 * ACT);
  unsigned short* Kb   = (unsigned short*)(ws + 4 * WW + 4 * ACT);
  unsigned short* Vt   = (unsigned short*)(ws + 4 * WW + 5 * ACT);
  unsigned short* Ctx  = Acts;  // activations are dead once QKV GEMMs finish

  cvt_weights_k<<<dim3(D_ * D_ / 256, 4), 256, 0, stream>>>(Wq, Wk, Wv, Wo, Wsw);
  cvt_act_k<<<dim3(B_ * S_ * D_ / (256 * 8), 3), 256, 0, stream>>>(query, key, value, Acts);

  const size_t AE = (size_t)B_ * S_ * D_;   // elements per activation tensor
  const float qscale = 0.125f * 1.44269504088896340736f;  // 1/sqrt(DH) * log2(e)
  dim3 gg(B_ * S_ / 64, D_ / 128);
  gemm_wmma_k<0><<<gg, 256, 0, stream>>>(Acts + 0 * AE, Wsw + 0 * (size_t)D_ * D_, bq, Qb, qscale);
  gemm_wmma_k<0><<<gg, 256, 0, stream>>>(Acts + 1 * AE, Wsw + 1 * (size_t)D_ * D_, bk, Kb, 1.0f);
  gemm_wmma_k<1><<<gg, 256, 0, stream>>>(Acts + 2 * AE, Wsw + 2 * (size_t)D_ * D_, bv, Vt, 1.0f);

  attention_k<<<dim3(S_ / 128, B_ * H_), 256, 0, stream>>>(Qb, Kb, Vt, Ctx);

  gemm_wmma_k<2><<<gg, 256, 0, stream>>>(Ctx, Wsw + 3 * (size_t)D_ * D_, bo, d_out, 1.0f);
}
